// MultiHeadAttention_40209483825386
// MI455X (gfx1250) — matrix operand order repaired and checked by reading, not compiled, not run on hardware
//
#include <hip/hip_runtime.h>
#include <hip/hip_bf16.h>

#define NHEADS 8
#define BATCH  4
#define NTOK   2048
#define DIN    512
#define DKEY   64
#define EMB    512

typedef __attribute__((ext_vector_type(16))) __bf16 v16bf;
typedef __attribute__((ext_vector_type(8)))  __bf16 v8bf;
typedef __attribute__((ext_vector_type(8)))  float  v8f;

#define WMMA_BF16(a, b, c) \
  __builtin_amdgcn_wmma_f32_16x16x32_bf16(false, (a), false, (b), (short)0, (c), false, false)

// ---------------------------------------------------------------------------
// Fragment loaders implementing the CDNA5 ISA 16-bit WMMA VGPR layouts.
// A (16x32, MxK): lane&15 = row M.  lo lanes: elems 0..7 -> K=0..7, 8..15 -> K=16..23
//                                   hi lanes: elems 0..7 -> K=8..15, 8..15 -> K=24..31
// B (32x16, KxN): lane&15 = col N.  same K mapping as A (lo: 0..7,16..23; hi: 8..15,24..31)
// C/D (16x16 f32): elem r -> row r + 8*(lane>=16), col lane&15
// ---------------------------------------------------------------------------
static __device__ __forceinline__ v16bf load_a_frag(const __bf16* base, int stride, int lane) {
  const int m  = lane & 15;
  const int hi = lane >> 4;
  const __bf16* p = base + (size_t)m * stride + hi * 8;
  v8bf lo = *(const v8bf*)(p);
  v8bf hc = *(const v8bf*)(p + 16);
  v16bf a;
#pragma unroll
  for (int i = 0; i < 8; ++i) { a[i] = lo[i]; a[i + 8] = hc[i]; }
  return a;
}

static __device__ __forceinline__ v16bf load_b_frag(const __bf16* base, int stride, int lane) {
  const int n  = lane & 15;
  const int hi = lane >> 4;
  const __bf16* p = base + (size_t)n * stride + hi * 8;
  v8bf b0 = *(const v8bf*)(p);
  v8bf b1 = *(const v8bf*)(p + 16);
  v16bf b;
#pragma unroll
  for (int i = 0; i < 8; ++i) { b[i] = b0[i]; b[i + 8] = b1[i]; }
  return b;
}

// ---------------------------------------------------------------------------
// Conversion / transpose helpers (negligible cost vs. the matmuls)
// ---------------------------------------------------------------------------
__global__ void cvt_f32_to_bf16(const float* __restrict__ src, __bf16* __restrict__ dst, int n) {
  for (int i = blockIdx.x * blockDim.x + threadIdx.x; i < n; i += gridDim.x * blockDim.x)
    dst[i] = (__bf16)src[i];
}

// src [batch][rows][cols] fp32  ->  dst [batch][cols][rows] bf16
__global__ void transpose_to_bf16(const float* __restrict__ src, __bf16* __restrict__ dst,
                                  int batch, int rows, int cols) {
  int total = batch * rows * cols;
  for (int i = blockIdx.x * blockDim.x + threadIdx.x; i < total; i += gridDim.x * blockDim.x) {
    int b = i / (rows * cols);
    int r = (i / cols) % rows;
    int c = i % cols;
    dst[(size_t)b * rows * cols + (size_t)c * rows + r] = (__bf16)src[i];
  }
}

// ---------------------------------------------------------------------------
// Kernel 1: QKV projections.  grid = (H*B*N/16, 3), block = 128 (4 waves).
// Each wave: one 16-token x 16-col tile, 16 chained WMMAs over DIN=512.
// Q is pre-scaled by 1/sqrt(64); V is stored transposed [h][b][v][token].
// ---------------------------------------------------------------------------
__global__ void __launch_bounds__(128)
proj_kernel(const __bf16* __restrict__ qbf, const __bf16* __restrict__ Wt_all,
            __bf16* __restrict__ Qb, __bf16* __restrict__ Kb, __bf16* __restrict__ Vt) {
  const int lane = threadIdx.x & 31;
  const int wave = threadIdx.x >> 5;
  const int proj = blockIdx.y;
  const int nt   = NTOK / 16;
  const int id   = blockIdx.x;
  const int h    = id / (BATCH * nt);
  const int rem  = id % (BATCH * nt);
  const int b    = rem / nt;
  const int n0   = (rem % nt) * 16;
  const int col0 = wave * 16;

  const __bf16* Abase = qbf + ((size_t)b * NTOK + n0) * DIN;
  const __bf16* Bbase = Wt_all + (((size_t)proj * NHEADS + h) * DKEY + col0) * DIN;

  v8f acc = {};
#pragma unroll
  for (int d0 = 0; d0 < DIN; d0 += 32) {
    v16bf a  = load_a_frag(Abase + d0, DIN, lane);
    v16bf bm = load_b_frag(Bbase + d0, DIN, lane);
    acc = WMMA_BF16(a, bm, acc);
  }

  const int hi = lane >> 4;
  const int nn = lane & 15;
  if (proj == 0) {
#pragma unroll
    for (int r = 0; r < 8; ++r) {
      int tok = n0 + r + hi * 8;
      Qb[(((size_t)(h * BATCH + b)) * NTOK + tok) * DKEY + col0 + nn] = (__bf16)(acc[r] * 0.125f);
    }
  } else if (proj == 1) {
#pragma unroll
    for (int r = 0; r < 8; ++r) {
      int tok = n0 + r + hi * 8;
      Kb[(((size_t)(h * BATCH + b)) * NTOK + tok) * DKEY + col0 + nn] = (__bf16)acc[r];
    }
  } else {
#pragma unroll
    for (int r = 0; r < 8; ++r) {
      int tok = n0 + r + hi * 8;
      int v   = col0 + nn;
      Vt[(((size_t)(h * BATCH + b)) * DKEY + v) * NTOK + tok] = (__bf16)acc[r];
    }
  }
}

// ---------------------------------------------------------------------------
// Kernel 2: flash attention.  grid = H*B*(N/16)/4, block = 128 (4 waves,
// each wave owns one 16-query tile).  Online softmax over 32-key chunks.
// ---------------------------------------------------------------------------
__global__ void __launch_bounds__(128)
attn_kernel(const __bf16* __restrict__ Qb, const __bf16* __restrict__ Kb,
            const __bf16* __restrict__ Vt, const int* __restrict__ mask,
            __bf16* __restrict__ Hcat) {
  __shared__ __bf16 Pl[4][16 * 32];

  const int lane = threadIdx.x & 31;
  const int wave = threadIdx.x >> 5;
  const int hi   = lane >> 4;
  const int nn   = lane & 15;
  const int nt   = NTOK / 16;
  const int tile = blockIdx.x * 4 + wave;
  const int h    = tile / (BATCH * nt);
  const int rem  = tile % (BATCH * nt);
  const int b    = rem / nt;
  const int n0   = (rem % nt) * 16;

  const float NEG_INF = -__builtin_inff();

  const __bf16* Qrow = Qb + (((size_t)(h * BATCH + b)) * NTOK + n0) * DKEY;
  const v16bf qa0 = load_a_frag(Qrow, DKEY, lane);
  const v16bf qa1 = load_a_frag(Qrow + 32, DKEY, lane);

  const __bf16* Kbase = Kb + ((size_t)(h * BATCH + b)) * NTOK * DKEY;
  const __bf16* Vbase = Vt + ((size_t)(h * BATCH + b)) * DKEY * NTOK;
  const int*    mbase = mask + ((size_t)b * NTOK + n0 + hi * 8) * NTOK + nn;

  float mrow[8], lrow[8];
  v8f O0 = {}, O1 = {}, O2 = {}, O3 = {};
#pragma unroll
  for (int r = 0; r < 8; ++r) { mrow[r] = NEG_INF; lrow[r] = 0.f; }

  __bf16* Pw = &Pl[wave][0];

  for (int g0 = 0; g0 < NTOK; g0 += 32) {
    if (g0 + 32 < NTOK)
      __builtin_prefetch(Kbase + (size_t)(g0 + 32) * DKEY, 0, 1);

    // ---- scores: two 16q x 16g tiles, dk=64 contracted in two K=32 steps
    v8f S0 = {}, S1 = {};
    {
      v16bf kb;
      kb = load_b_frag(Kbase + (size_t)g0 * DKEY, DKEY, lane);
      S0 = WMMA_BF16(qa0, kb, S0);
      kb = load_b_frag(Kbase + (size_t)g0 * DKEY + 32, DKEY, lane);
      S0 = WMMA_BF16(qa1, kb, S0);
      kb = load_b_frag(Kbase + (size_t)(g0 + 16) * DKEY, DKEY, lane);
      S1 = WMMA_BF16(qa0, kb, S1);
      kb = load_b_frag(Kbase + (size_t)(g0 + 16) * DKEY + 32, DKEY, lane);
      S1 = WMMA_BF16(qa1, kb, S1);
    }

    // ---- mask to -inf, row max over 32 keys (cross-lane within 16-lane group)
    float p0[8], p1[8], red[8];
#pragma unroll
    for (int r = 0; r < 8; ++r) {
      int m0 = mbase[(size_t)r * NTOK + g0];
      int m1 = mbase[(size_t)r * NTOK + g0 + 16];
      p0[r] = m0 ? NEG_INF : S0[r];
      p1[r] = m1 ? NEG_INF : S1[r];
      red[r] = fmaxf(p0[r], p1[r]);
    }
#pragma unroll
    for (int off = 8; off; off >>= 1)
#pragma unroll
      for (int r = 0; r < 8; ++r)
        red[r] = fmaxf(red[r], __shfl_xor(red[r], off, 32));

    // ---- online softmax update (guard fully-masked rows: no NaNs, output 0)
    float corr[8];
#pragma unroll
    for (int r = 0; r < 8; ++r) {
      float mn   = fmaxf(mrow[r], red[r]);
      bool  dead = (mn == NEG_INF);
      corr[r] = dead ? 1.f : __expf(mrow[r] - mn);
      p0[r]   = dead ? 0.f : __expf(p0[r] - mn);
      p1[r]   = dead ? 0.f : __expf(p1[r] - mn);
      mrow[r] = mn;
      red[r]  = p0[r] + p1[r];
    }
#pragma unroll
    for (int off = 8; off; off >>= 1)
#pragma unroll
      for (int r = 0; r < 8; ++r)
        red[r] += __shfl_xor(red[r], off, 32);

#pragma unroll
    for (int r = 0; r < 8; ++r) {
      lrow[r] = lrow[r] * corr[r] + red[r];
      O0[r] *= corr[r]; O1[r] *= corr[r]; O2[r] *= corr[r]; O3[r] *= corr[r];
      int row = r + hi * 8;                 // C-layout row for this element
      Pw[row * 32 + nn]      = (__bf16)p0[r];
      Pw[row * 32 + 16 + nn] = (__bf16)p1[r];
    }

    // ---- P (16x32) via LDS into A-layout, then 4 PV WMMAs (same-wave DS is in-order)
    v16bf pa = load_a_frag(Pw, 32, lane);
    v16bf vb;
    vb = load_b_frag(Vbase + (size_t)0 * 16 * NTOK + g0, NTOK, lane);
    O0 = WMMA_BF16(pa, vb, O0);
    vb = load_b_frag(Vbase + (size_t)1 * 16 * NTOK + g0, NTOK, lane);
    O1 = WMMA_BF16(pa, vb, O1);
    vb = load_b_frag(Vbase + (size_t)2 * 16 * NTOK + g0, NTOK, lane);
    O2 = WMMA_BF16(pa, vb, O2);
    vb = load_b_frag(Vbase + (size_t)3 * 16 * NTOK + g0, NTOK, lane);
    O3 = WMMA_BF16(pa, vb, O3);
  }

  // ---- normalize and emit heads concatenated: Hcat[b][tok][h*64 + v]
#pragma unroll
  for (int r = 0; r < 8; ++r) {
    int   tok  = n0 + r + hi * 8;
    float invl = (lrow[r] > 0.f) ? 1.f / lrow[r] : 0.f;
    size_t o = ((size_t)b * NTOK + tok) * EMB + h * DKEY + nn;
    Hcat[o + 0]  = (__bf16)(O0[r] * invl);
    Hcat[o + 16] = (__bf16)(O1[r] * invl);
    Hcat[o + 32] = (__bf16)(O2[r] * invl);
    Hcat[o + 48] = (__bf16)(O3[r] * invl);
  }
}

// ---------------------------------------------------------------------------
// Kernel 3: output projection, (B*N=8192) x 512 x 512 bf16 GEMM -> fp32 out.
// grid = 512 row-tiles * 8 col-blocks, block = 128 (4 waves x 16 cols).
// ---------------------------------------------------------------------------
__global__ void __launch_bounds__(128)
outproj_kernel(const __bf16* __restrict__ Hcat, const __bf16* __restrict__ WoT,
               float* __restrict__ out) {
  const int lane = threadIdx.x & 31;
  const int wave = threadIdx.x >> 5;
  const int id   = blockIdx.x;
  const int row0 = (id >> 3) * 16;
  const int col0 = (id & 7) * 64 + wave * 16;

  const __bf16* Abase = Hcat + (size_t)row0 * EMB;
  const __bf16* Bbase = WoT + (size_t)col0 * EMB;

  v8f acc = {};
#pragma unroll
  for (int d0 = 0; d0 < EMB; d0 += 32) {
    v16bf a  = load_a_frag(Abase + d0, EMB, lane);
    v16bf bm = load_b_frag(Bbase + d0, EMB, lane);
    acc = WMMA_BF16(a, bm, acc);
  }

  const int hi = lane >> 4;
  const int nn = lane & 15;
#pragma unroll
  for (int r = 0; r < 8; ++r)
    out[((size_t)row0 + r + hi * 8) * EMB + col0 + nn] = acc[r];
}

// ---------------------------------------------------------------------------
extern "C" void kernel_launch(void* const* d_in, const int* in_sizes, int n_in,
                              void* d_out, int out_size, void* d_ws, size_t ws_size,
                              hipStream_t stream) {
  (void)in_sizes; (void)n_in; (void)out_size; (void)ws_size;

  const float* q    = (const float*)d_in[0];
  const int*   mask = (const int*)d_in[1];
  const float* Wq   = (const float*)d_in[2];
  const float* Wk   = (const float*)d_in[3];
  const float* Wv   = (const float*)d_in[4];
  const float* Wo   = (const float*)d_in[5];
  float*       out  = (float*)d_out;

  char*  ws  = (char*)d_ws;
  size_t off = 0;
  auto alloc = [&](size_t bytes) -> char* {
    char* p = ws + off;
    off += (bytes + 255) & ~(size_t)255;
    return p;
  };

  __bf16* qbf  = (__bf16*)alloc((size_t)BATCH * NTOK * DIN * 2);
  __bf16* WtA  = (__bf16*)alloc((size_t)3 * NHEADS * DKEY * DIN * 2);  // [3][h][k][d]
  __bf16* WoT  = (__bf16*)alloc((size_t)EMB * EMB * 2);                // [e][d]
  __bf16* Qb   = (__bf16*)alloc((size_t)NHEADS * BATCH * NTOK * DKEY * 2);
  __bf16* Kb   = (__bf16*)alloc((size_t)NHEADS * BATCH * NTOK * DKEY * 2);
  __bf16* Vt   = (__bf16*)alloc((size_t)NHEADS * BATCH * DKEY * NTOK * 2);
  __bf16* Hcat = (__bf16*)alloc((size_t)BATCH * NTOK * EMB * 2);

  const size_t wstride = (size_t)NHEADS * DKEY * DIN;

  cvt_f32_to_bf16<<<2048, 256, 0, stream>>>(q, qbf, BATCH * NTOK * DIN);
  transpose_to_bf16<<<512, 256, 0, stream>>>(Wq, WtA + 0 * wstride, NHEADS, DIN, DKEY);
  transpose_to_bf16<<<512, 256, 0, stream>>>(Wk, WtA + 1 * wstride, NHEADS, DIN, DKEY);
  transpose_to_bf16<<<512, 256, 0, stream>>>(Wv, WtA + 2 * wstride, NHEADS, DIN, DKEY);
  transpose_to_bf16<<<512, 256, 0, stream>>>(Wo, WoT, 1, EMB, EMB);

  proj_kernel<<<dim3(NHEADS * BATCH * (NTOK / 16), 3), 128, 0, stream>>>(qbf, WtA, Qb, Kb, Vt);
  attn_kernel<<<NHEADS * BATCH * (NTOK / 16) / 4, 128, 0, stream>>>(Qb, Kb, Vt, mask, Hcat);
  outproj_kernel<<<(BATCH * NTOK / 16) * (EMB / 64), 128, 0, stream>>>(Hcat, WoT, out);
}
